// Attention_81690277970645
// MI455X (gfx1250) — matrix operand order repaired and checked by reading, compiled, not run on hardware
//
#include <hip/hip_runtime.h>

typedef __attribute__((ext_vector_type(16))) _Float16 v16h;
typedef __attribute__((ext_vector_type(8)))  _Float16 v8h;
typedef __attribute__((ext_vector_type(8)))  float    v8f;

#define HEADS 16
#define DHEAD 64
#define NSEQ  2048
#define NROWS 4096            // b*n = 2*2048
#define SCALE 0.125f          // 64^-0.5

// workspace layout, in _Float16 units (total 25,165,824 f16 = 48 MB)
#define OFF_XH  0L
#define OFF_WQ  (OFF_XH + 4096L*1024)
#define OFF_WK  (OFF_WQ + 1024L*1024)
#define OFF_WV  (OFF_WK + 1024L*1024)
#define OFF_WO  (OFF_WV + 1024L*1024)
#define OFF_Q   (OFF_WO + 1024L*1024)   // [b,h,n,d]
#define OFF_K   (OFF_Q  + 4096L*1024)   // [b,h,n,d]
#define OFF_VT  (OFF_K  + 4096L*1024)   // [b,h,d,n]  (transposed)
#define OFF_AO  (OFF_VT + 4096L*1024)   // [b*n, 1024]

union H16 { v16h v; struct { v8h lo, hi; } p; };

__device__ __forceinline__ v8f wmma16(v16h a, v16h b, v8f c) {
  return __builtin_amdgcn_wmma_f32_16x16x32_f16(false, a, false, b,
                                                (short)0, c, false, false);
}

// ------------------------------------------------- Tensor Data Mover staging
#if __has_builtin(__builtin_amdgcn_tensor_load_to_lds)
#define HAS_TDM 1
typedef __attribute__((ext_vector_type(4))) unsigned int u32x4;
typedef __attribute__((ext_vector_type(8))) int i32x8;
typedef __attribute__((ext_vector_type(4))) int i32x4;

// 2D tile DMA: global (row-major, stride s0 elements of 2B) -> LDS with
// row padding (pad_interval/pad_amount codes per CDNA5 D# spec, 4-dword pad).
__device__ __forceinline__ void tdm_load_2d(unsigned lds_off, const void* g,
                                            unsigned td0, unsigned td1,
                                            unsigned long s0,
                                            unsigned tl0, unsigned tl1,
                                            unsigned pint, unsigned pamt) {
  unsigned long ga = (unsigned long)g;
  u32x4 g0;
  g0[0] = 1u;                                     // count=1, user descriptor
  g0[1] = lds_off;                                // lds_addr (bytes)
  g0[2] = (unsigned)ga;                           // global_addr[31:0]
  g0[3] = (unsigned)((ga >> 32) & 0x01ffffffu) |  // global_addr[56:32]
          0x80000000u;                            // type=2 ("image")
  i32x8 g1;
  g1[0] = (int)((1u << 16) |                      // data_size = 2 bytes
                (1u << 20) |                      // pad_enable
                (pint << 22) | (pamt << 25));     // pad interval/amount codes
  g1[1] = (int)((td0 & 0xffffu) << 16);           // tensor_dim0[15:0]
  g1[2] = (int)(((td0 >> 16) & 0xffffu) | ((td1 & 0xffffu) << 16));
  g1[3] = (int)(((td1 >> 16) & 0xffffu) | ((tl0 & 0xffffu) << 16));
  g1[4] = (int)(tl1 & 0xffffu);                   // tile_dim1 (tile_dim2=0)
  g1[5] = (int)(unsigned)s0;                      // tensor_dim0_stride[31:0]
  g1[6] = (int)(unsigned)((s0 >> 32) & 0xffffu);  // stride[47:32]
  g1[7] = 0;
  i32x4 z4 = {0, 0, 0, 0};
#if defined(__clang_major__) && (__clang_major__ >= 23)
  i32x8 z8 = {0, 0, 0, 0, 0, 0, 0, 0};
  __builtin_amdgcn_tensor_load_to_lds(g0, g1, z4, z4, z8, 0);  // 6-arg form
#else
  __builtin_amdgcn_tensor_load_to_lds(g0, g1, z4, z4, 0);      // 5-arg form
#endif
}
#else
#define HAS_TDM 0
#endif

// ---------------------------------------------------------------- convert
__global__ __launch_bounds__(256)
void cvt_kernel(const float* __restrict__ x,  const float* __restrict__ wq,
                const float* __restrict__ wk, const float* __restrict__ wv,
                const float* __restrict__ wo, _Float16* __restrict__ ws) {
  long i = (long)blockIdx.x * 256 + threadIdx.x;
  const long NX = 4096L * 1024, NW = 1024L * 1024;
  if (i < NX) { ws[OFF_XH + i] = (_Float16)x[i]; return; }
  long j = i - NX;
  if (j < NW) { ws[OFF_WQ + j] = (_Float16)wq[j]; return; }
  j -= NW;
  if (j < NW) { ws[OFF_WK + j] = (_Float16)wk[j]; return; }
  j -= NW;
  if (j < NW) { ws[OFF_WV + j] = (_Float16)wv[j]; return; }
  j -= NW;
  if (j < NW) { ws[OFF_WO + j] = (_Float16)wo[j]; }
}

// ---------------------------------------------------------------- fused QKV GEMM
// grid: (16 M-tiles of 256 rows, 48); blockIdx.y: /16 -> {Q,K,V}, %16 -> col tile
__global__ __launch_bounds__(256)
void qkv_gemm(_Float16* __restrict__ ws, const float* __restrict__ bq,
              const float* __restrict__ bk, const float* __restrict__ bv) {
  __shared__ __align__(16) _Float16 At[256 * 40];   // [m][k], stride 40 (16dw+4pad)
  __shared__ __align__(16) _Float16 Bt[64 * 40];    // [n][k] transposed
  const _Float16* xh = ws + OFF_XH;
  const int m0  = blockIdx.x * 256;
  const int sel = blockIdx.y >> 4;
  const int n0  = (blockIdx.y & 15) * 64;
  const _Float16* wh = ws + (sel == 0 ? OFF_WQ : sel == 1 ? OFF_WK : OFF_WV);
  const float* bias  = (sel == 0 ? bq : sel == 1 ? bk : bv);
  const int t = threadIdx.x, w = t >> 5, lane = t & 31;
  const int half = lane >> 4, l16 = lane & 15;

  v8f acc[2][4] = {};
  for (int k0 = 0; k0 < 1024; k0 += 32) {
    __syncthreads();
#if HAS_TDM
    if (w == 0)   // TDM: 256x32 f16 tile, LDS rows padded to 40 f16
      tdm_load_2d((unsigned)(size_t)At, xh + (long)m0 * 1024 + k0,
                  32, 256, 1024, 32, 256, /*pint 16dw*/3, /*pamt 4dw*/3);
#else
    #pragma unroll
    for (int i = 0; i < 4; ++i) {
      int f  = (t * 4 + i) * 8;
      int mm = f >> 5, cc = f & 31;
      *(uint4*)(At + mm * 40 + cc) =
          *(const uint4*)(xh + (long)(m0 + mm) * 1024 + k0 + cc);
    }
#endif
    { // B tile 32x64 -> transposed [n][k]: coalesced 16B read + LDS scatter
      int k = t >> 3, nch = t & 7;
      v8h vb = *(const v8h*)(wh + (long)(k0 + k) * 1024 + n0 + nch * 8);
      #pragma unroll
      for (int i = 0; i < 8; ++i) Bt[(nch * 8 + i) * 40 + k] = vb[i];
    }
#if HAS_TDM
    if (w == 0) __builtin_amdgcn_s_wait_tensorcnt(0);
#endif
    __syncthreads();
    H16 a[2];
    #pragma unroll
    for (int st = 0; st < 2; ++st) {
      const _Float16* ap = At + (w * 32 + st * 16 + l16) * 40;
      a[st].p.lo = *(const v8h*)(ap + half * 8);
      a[st].p.hi = *(const v8h*)(ap + 16 + half * 8);
    }
    #pragma unroll
    for (int tn = 0; tn < 4; ++tn) {
      H16 b;
      const _Float16* bp = Bt + (tn * 16 + l16) * 40 + 8 * half;
      b.p.lo = *(const v8h*)bp;
      b.p.hi = *(const v8h*)(bp + 16);
      #pragma unroll
      for (int st = 0; st < 2; ++st)
        acc[st][tn] = wmma16(a[st].v, b.v, acc[st][tn]);
    }
  }
  // epilogue: bias + layout conversion into Q/K [b,h,n,d] or V^T [b,h,d,n]
  _Float16* Qb = ws + OFF_Q;
  _Float16* Kb = ws + OFF_K;
  _Float16* Vt = ws + OFF_VT;
  #pragma unroll
  for (int tn = 0; tn < 4; ++tn) {
    int c = n0 + tn * 16 + l16;
    int h = c >> 6, d = c & 63;
    float bi = bias[c];
    #pragma unroll
    for (int st = 0; st < 2; ++st)
      #pragma unroll
      for (int j = 0; j < 8; ++j) {
        int r = m0 + w * 32 + st * 16 + j + 8 * half;
        int b_ = r >> 11, nseq = r & 2047;
        _Float16 vv = (_Float16)(acc[st][tn][j] + bi);
        long bh = (long)(b_ * HEADS + h);
        if (sel == 0)      Qb[(bh * 2048 + nseq) * 64 + d] = vv;
        else if (sel == 1) Kb[(bh * 2048 + nseq) * 64 + d] = vv;
        else               Vt[(bh * 64 + d) * 2048 + nseq] = vv;
      }
  }
}

// ---------------------------------------------------------------- flash attention
// grid: (b*h = 32, 2048/128 = 16 query tiles), 8 waves, 16 query rows/wave
__global__ __launch_bounds__(256)
void attn_kernel(_Float16* __restrict__ ws) {
  __shared__ __align__(16) _Float16 Kblk[32 * 72];   // [key][d], 32dw + 4dw pad
  __shared__ __align__(16) _Float16 Vblk[64 * 40];   // [d][key], 16dw + 4dw pad
  __shared__ __align__(16) _Float16 Pst[8 * 16 * 32];// per-wave 16x32 P staging
  const _Float16* Qb = ws + OFF_Q;
  const _Float16* Kb = ws + OFF_K;
  const _Float16* Vt = ws + OFF_VT;
  _Float16* AO = ws + OFF_AO;
  const int bh = blockIdx.x, qt = blockIdx.y;
  const int t = threadIdx.x, w = t >> 5, lane = t & 31;
  const int half = lane >> 4, l16 = lane & 15;
  const int q0 = qt * 128 + w * 16;

  H16 qa[2];
  {
    long rowq = (long)bh * 2048 + q0 + l16;
    #pragma unroll
    for (int kc = 0; kc < 2; ++kc) {
      qa[kc].p.lo = *(const v8h*)(Qb + rowq * 64 + kc * 32 + half * 8);
      qa[kc].p.hi = *(const v8h*)(Qb + rowq * 64 + kc * 32 + 16 + half * 8);
    }
  }
  float mi[8], li[8];
  #pragma unroll
  for (int j = 0; j < 8; ++j) { mi[j] = -1e30f; li[j] = 0.f; }
  v8f o[4] = {};

  for (int kb = 0; kb < 2048; kb += 32) {
    __syncthreads();
#if HAS_TDM
    if (w == 0) { // DMA both tiles; pads reproduce the LDS strides (72 / 40)
      tdm_load_2d((unsigned)(size_t)Kblk,
                  Kb + ((long)bh * 2048 + kb) * 64,
                  64, 32, 64, 64, 32, /*pint 32dw*/4, /*pamt 4dw*/3);
      tdm_load_2d((unsigned)(size_t)Vblk,
                  Vt + (long)bh * 64 * 2048 + kb,
                  32, 64, 2048, 32, 64, /*pint 16dw*/3, /*pamt 4dw*/3);
      __builtin_amdgcn_s_wait_tensorcnt(0);
    }
#else
    {
      int kk = t >> 3, ch = t & 7;
      *(uint4*)(Kblk + kk * 72 + ch * 8) =
          *(const uint4*)(Kb + ((long)bh * 2048 + kb + kk) * 64 + ch * 8);
      int d = t >> 2, c2 = t & 3;
      *(uint4*)(Vblk + d * 40 + c2 * 8) =
          *(const uint4*)(Vt + ((long)bh * 64 + d) * 2048 + kb + c2 * 8);
    }
#endif
    __syncthreads();

    // S = Q * K^T  (two 16x16 key tiles, d=64 => 2 WMMAs each)
    v8f s0 = {}, s1 = {};
    #pragma unroll
    for (int kc = 0; kc < 2; ++kc) {
      H16 b0, b1;
      const _Float16* p0 = Kblk + l16 * 72 + kc * 32 + 8 * half;
      const _Float16* p1 = Kblk + (16 + l16) * 72 + kc * 32 + 8 * half;
      b0.p.lo = *(const v8h*)p0;  b0.p.hi = *(const v8h*)(p0 + 16);
      b1.p.lo = *(const v8h*)p1;  b1.p.hi = *(const v8h*)(p1 + 16);
      s0 = wmma16(qa[kc].v, b0.v, s0);
      s1 = wmma16(qa[kc].v, b1.v, s1);
    }

    // online softmax: rows j+8*half live across the 16 lanes of each half-group
    float al[8];
    #pragma unroll
    for (int j = 0; j < 8; ++j) {
      float v0 = s0[j] * SCALE, v1 = s1[j] * SCALE;
      float mx = fmaxf(v0, v1);
      #pragma unroll
      for (int mk = 1; mk < 16; mk <<= 1) mx = fmaxf(mx, __shfl_xor(mx, mk, 32));
      float mn = fmaxf(mi[j], mx);
      float a  = __expf(mi[j] - mn);
      float p0 = __expf(v0 - mn);
      float p1 = __expf(v1 - mn);
      float rs = p0 + p1;
      #pragma unroll
      for (int mk = 1; mk < 16; mk <<= 1) rs += __shfl_xor(rs, mk, 32);
      mi[j] = mn;  li[j] = li[j] * a + rs;  al[j] = a;
      Pst[w * 512 + (j + 8 * half) * 32 + l16]      = (_Float16)p0;
      Pst[w * 512 + (j + 8 * half) * 32 + 16 + l16] = (_Float16)p1;
    }
    #pragma unroll
    for (int tn = 0; tn < 4; ++tn)
      #pragma unroll
      for (int j = 0; j < 8; ++j) o[tn][j] *= al[j];

    // reload P as A-fragment (wave-private region; LDS ops in-order per wave)
    H16 pa;
    pa.p.lo = *(const v8h*)(Pst + w * 512 + l16 * 32 + half * 8);
    pa.p.hi = *(const v8h*)(Pst + w * 512 + l16 * 32 + 16 + half * 8);

    // O += P * V  (4 d-tiles of 16)
    #pragma unroll
    for (int tn = 0; tn < 4; ++tn) {
      H16 b;
      const _Float16* vp = Vblk + (tn * 16 + l16) * 40 + 8 * half;
      b.p.lo = *(const v8h*)vp;  b.p.hi = *(const v8h*)(vp + 16);
      o[tn] = wmma16(pa.v, b.v, o[tn]);
    }
  }

  // finalize: divide by row sums, write AO [b*n, 1024] f16
  const int b_ = bh >> 4, h = bh & 15;
  #pragma unroll
  for (int tn = 0; tn < 4; ++tn)
    #pragma unroll
    for (int j = 0; j < 8; ++j) {
      int nseq = q0 + j + 8 * half;
      int col  = h * 64 + tn * 16 + l16;
      AO[(long)(b_ * 2048 + nseq) * 1024 + col] = (_Float16)(o[tn][j] / li[j]);
    }
}

// ---------------------------------------------------------------- output GEMM
__global__ __launch_bounds__(256)
void out_gemm(const _Float16* __restrict__ ws, const float* __restrict__ bo,
              float* __restrict__ out) {
  __shared__ __align__(16) _Float16 At[256 * 40];
  __shared__ __align__(16) _Float16 Bt[64 * 40];
  const _Float16* ah = ws + OFF_AO;
  const _Float16* wh = ws + OFF_WO;
  const int m0 = blockIdx.x * 256, n0 = blockIdx.y * 64;
  const int t = threadIdx.x, w = t >> 5, lane = t & 31;
  const int half = lane >> 4, l16 = lane & 15;

  v8f acc[2][4] = {};
  for (int k0 = 0; k0 < 1024; k0 += 32) {
    __syncthreads();
#if HAS_TDM
    if (w == 0)
      tdm_load_2d((unsigned)(size_t)At, ah + (long)m0 * 1024 + k0,
                  32, 256, 1024, 32, 256, 3, 3);
#else
    #pragma unroll
    for (int i = 0; i < 4; ++i) {
      int f  = (t * 4 + i) * 8;
      int mm = f >> 5, cc = f & 31;
      *(uint4*)(At + mm * 40 + cc) =
          *(const uint4*)(ah + (long)(m0 + mm) * 1024 + k0 + cc);
    }
#endif
    {
      int k = t >> 3, nch = t & 7;
      v8h vb = *(const v8h*)(wh + (long)(k0 + k) * 1024 + n0 + nch * 8);
      #pragma unroll
      for (int i = 0; i < 8; ++i) Bt[(nch * 8 + i) * 40 + k] = vb[i];
    }
#if HAS_TDM
    if (w == 0) __builtin_amdgcn_s_wait_tensorcnt(0);
#endif
    __syncthreads();
    H16 a[2];
    #pragma unroll
    for (int st = 0; st < 2; ++st) {
      const _Float16* ap = At + (w * 32 + st * 16 + l16) * 40;
      a[st].p.lo = *(const v8h*)(ap + half * 8);
      a[st].p.hi = *(const v8h*)(ap + 16 + half * 8);
    }
    #pragma unroll
    for (int tn = 0; tn < 4; ++tn) {
      H16 b;
      const _Float16* bp = Bt + (tn * 16 + l16) * 40 + 8 * half;
      b.p.lo = *(const v8h*)bp;  b.p.hi = *(const v8h*)(bp + 16);
      #pragma unroll
      for (int st = 0; st < 2; ++st)
        acc[st][tn] = wmma16(a[st].v, b.v, acc[st][tn]);
    }
  }
  #pragma unroll
  for (int tn = 0; tn < 4; ++tn) {
    int c = n0 + tn * 16 + l16;
    float bi = bo[c];
    #pragma unroll
    for (int st = 0; st < 2; ++st)
      #pragma unroll
      for (int j = 0; j < 8; ++j) {
        int r = m0 + w * 32 + st * 16 + j + 8 * half;
        out[(long)r * 1024 + c] = acc[st][tn][j] + bi;
      }
  }
}

// ---------------------------------------------------------------- launch
extern "C" void kernel_launch(void* const* d_in, const int* in_sizes, int n_in,
                              void* d_out, int out_size, void* d_ws, size_t ws_size,
                              hipStream_t stream) {
  const float* x  = (const float*)d_in[0];
  const float* wq = (const float*)d_in[1];
  const float* bq = (const float*)d_in[2];
  const float* wk = (const float*)d_in[3];
  const float* bk = (const float*)d_in[4];
  const float* wv = (const float*)d_in[5];
  const float* bv = (const float*)d_in[6];
  const float* wo = (const float*)d_in[7];
  const float* bo = (const float*)d_in[8];
  float* out = (float*)d_out;
  _Float16* ws = (_Float16*)d_ws;

  cvt_kernel<<<32768, 256, 0, stream>>>(x, wq, wk, wv, wo, ws);
  qkv_gemm<<<dim3(16, 48), 256, 0, stream>>>(ws, bq, bk, bv);
  attn_kernel<<<dim3(32, 16), 256, 0, stream>>>(ws);
  out_gemm<<<dim3(16, 16), 256, 0, stream>>>(ws, bo, out);
}
